// GraphModel_36283883717390
// MI455X (gfx1250) — matrix operand order repaired and checked by reading, not compiled, not run on hardware
//
#include <hip/hip_runtime.h>

#define EMB    768
#define HEADS  8
#define HD     96
#define NLAYER 3
#define FFD    3072
#define BATCH  32
#define NNODE  510
#define TLEN   512
#define MROWS  (BATCH*TLEN)

typedef unsigned short u16;
typedef unsigned int   u32;
typedef unsigned long long u64;
typedef __attribute__((ext_vector_type(16))) __bf16 v16bf;
typedef __attribute__((ext_vector_type(8)))  float  v8f;
typedef __attribute__((ext_vector_type(4)))  u32    tdm_v4u;
typedef __attribute__((ext_vector_type(8)))  int    tdm_v8i;
typedef __attribute__((ext_vector_type(4)))  int    tdm_v4i;

union FragB { v16bf v; u16 u[16]; };

__device__ inline u16 f2bf(float x) {
  u32 u = __float_as_uint(x);
  u = (u + 0x7FFFu + ((u >> 16) & 1u)) >> 16;
  return (u16)u;
}

// ---- WMMA fragment loaders (contiguous k = 16*(lane/16) + element, wave32) ----------------

// A: 16xK row-major bf16, row m in lane (l&15); lane half selects K half (16 contiguous).
__device__ inline v16bf load_a_frag(const u16* A, int m0, int kt, int ld) {
  int l  = threadIdx.x & 31;
  int m  = m0 + (l & 15);
  int kh = (l >> 4) * 16;
  const u16* base = A + (size_t)m * ld + kt + kh;
  FragB f;
#pragma unroll
  for (int i = 0; i < 4; ++i) {
    *(u32*)&f.u[2*i]     = *(const u32*)(base + 2*i);       // K = kt+kh+2i,2i+1
    *(u32*)&f.u[2*i + 8] = *(const u32*)(base + 8 + 2*i);   // K = kt+kh+8+2i
  }
  return f.v;
}

// B (Kx16) where column n is CONTIGUOUS along k in memory: B[k][n] = W[n*ldk + k]
__device__ inline v16bf load_b_frag_t(const u16* Bm, int n0, int kt, int ldk) {
  int l = threadIdx.x & 31;
  int n = n0 + (l & 15);
  const u16* base = Bm + (size_t)n * ldk + kt + (l >> 4) * 16;
  FragB f;
#pragma unroll
  for (int i = 0; i < 8; ++i)
    *(u32*)&f.u[2*i] = *(const u32*)(base + 2*i);
  return f.v;
}

// B (Kx16) plain row-major: B[k][n] = W[k*ldn + n]   (used only on LDS tiles)
__device__ inline v16bf load_b_frag_n(const u16* Bm, int n0, int kt, int ldn) {
  int l  = threadIdx.x & 31;
  int n  = n0 + (l & 15);
  int k0 = kt + (l >> 4) * 16;
  FragB f;
#pragma unroll
  for (int e = 0; e < 16; ++e)
    f.u[e] = Bm[(size_t)(k0 + e) * ldn + n];
  return f.v;
}

__device__ inline v8f wmma_bf16(v16bf a, v16bf b, v8f c) {
  return __builtin_amdgcn_wmma_f32_16x16x32_bf16(false, a, false, b, (short)0, c, false, false);
}

__device__ inline void store_acc_bf(u16* C, v8f acc, size_t row0, int n, int ld) {
  int kh = (threadIdx.x >> 4) & 1;
#pragma unroll
  for (int r = 0; r < 8; ++r)
    C[(row0 + r + 8*kh) * (size_t)ld + n] = f2bf(acc[r]);
}

// ---- Tensor Data Mover: 2-D bf16 tile load Global -> LDS (ISA ch.8 D#) ---
// tensor = (td1 lines) x (td0 elements/line), row pitch stride0 (elements);
// tile   = (tl1 lines) x (tl0 elements/line), packed into LDS at lds ptr.
__device__ inline void tdm_load_2d(const void* gsrc, const u16* lds,
                                   u32 td0, u32 td1, u32 tl0, u32 tl1,
                                   u64 stride0) {
  u64 ga = (u64)gsrc;
  u32 lo = (u32)(size_t)lds;         // low 32 bits of generic ptr = LDS offset
  tdm_v4u g0;
  g0[0] = 1u;                                            // count=1, user mode
  g0[1] = lo;                                            // lds_addr
  g0[2] = (u32)ga;                                       // global_addr[31:0]
  g0[3] = ((u32)(ga >> 32) & 0x01FFFFFFu) | 0x80000000u; // addr[56:32] | type=2
  tdm_v8i g1;
  g1[0] = (int)(1u << 16);                               // data_size = 2 bytes
  g1[1] = (int)((td0 & 0xFFFFu) << 16);                  // tensor_dim0[15:0]
  g1[2] = (int)((td0 >> 16) | ((td1 & 0xFFFFu) << 16));  // td0 hi | td1 lo
  g1[3] = (int)((td1 >> 16) | (tl0 << 16));              // td1 hi | tile_dim0
  g1[4] = (int)(tl1 & 0xFFFFu);                          // tile_dim1 (tile_dim2=0)
  g1[5] = (int)(u32)(stride0 & 0xFFFFFFFFu);             // dim0 stride lo
  g1[6] = (int)(u32)((stride0 >> 32) & 0xFFFFu);         // dim0 stride hi
  g1[7] = 0;
  tdm_v4i z4 = {0, 0, 0, 0};
  tdm_v8i z8 = {0, 0, 0, 0, 0, 0, 0, 0};
  __builtin_amdgcn_tensor_load_to_lds(g0, g1, z4, z4, z8, 0);
}

// ---- small utility kernels ---------------------------------------------

__global__ void cvt_bf16_kernel(const float* __restrict__ src, u16* __restrict__ dst, int n) {
  int i = blockIdx.x * 256 + threadIdx.x;
  if (i < n) dst[i] = f2bf(src[i]);
}

__global__ void build_x_kernel(const float* __restrict__ embs, const float* __restrict__ vs,
                               const float* __restrict__ vt, float* __restrict__ xf,
                               u16* __restrict__ xb) {
  size_t idx = (size_t)blockIdx.x * 256 + threadIdx.x;
  if (idx >= (size_t)MROWS * EMB) return;
  int e = (int)(idx % EMB);
  size_t bt = idx / EMB;
  int t = (int)(bt % TLEN);
  int b = (int)(bt / TLEN);
  float v = (t == 0) ? vs[e] : (t == 1) ? vt[e]
            : embs[((size_t)b * NNODE + (t - 2)) * EMB + e];
  xf[idx] = v;
  xb[idx] = f2bf(v);
}

// ---- QKV projection: one wave -> 16 rows x 96 cols ----------------------
// which==2 stores V transposed: vt[(bh*HD + d)*TLEN + t]  (contiguous in t)

__global__ void __launch_bounds__(32)
qkv_kernel(const u16* __restrict__ xb, const u16* __restrict__ wq,
           const u16* __restrict__ wk, const u16* __restrict__ wv,
           u16* __restrict__ qo, u16* __restrict__ ko, u16* __restrict__ vto) {
  int qt = blockIdx.x;            // T/16
  int bh = blockIdx.y;            // b*H + h
  int which = blockIdx.z;         // 0=q 1=k 2=v
  int h = bh % HEADS, b = bh / HEADS;
  const u16* W = (which == 0) ? wq : (which == 1) ? wk : wv;
  const u16* A = xb + h * HD;     // rows b*T+t, stride EMB
  int arow0 = b * TLEN + qt * 16;

  v8f acc[6];
#pragma unroll
  for (int j = 0; j < 6; ++j) acc[j] = (v8f){0,0,0,0,0,0,0,0};

  for (int kt = 0; kt < HD; kt += 32) {
    v16bf af = load_a_frag(A, arow0, kt, EMB);
#pragma unroll
    for (int j = 0; j < 6; ++j) {
      v16bf bf = load_b_frag_t(W, j * 16, kt, HD);   // q[e] = sum_d x[d]*W[e,d]
      acc[j] = wmma_bf16(af, bf, acc[j]);
    }
  }
  int lm = threadIdx.x & 15, kh = (threadIdx.x >> 4) & 1;
  if (which == 2) {
    // transposed store: lane owns column d = j*16+lm, rows t contiguous
#pragma unroll
    for (int j = 0; j < 6; ++j) {
      size_t basev = ((size_t)bh * HD + j * 16 + lm) * TLEN + qt * 16 + 8 * kh;
#pragma unroll
      for (int r = 0; r < 8; ++r) vto[basev + r] = f2bf(acc[j][r]);
    }
  } else {
    u16* out = (which == 0) ? qo : ko;
    size_t orow0 = (size_t)bh * TLEN + qt * 16;
#pragma unroll
    for (int j = 0; j < 6; ++j)
      store_acc_bf(out, acc[j], orow0, j * 16 + lm, HD);
  }
}

// ---- Attention: one wave per (b,h,16-row q tile) -------------------------

__global__ void __launch_bounds__(32)
attn_kernel(const u16* __restrict__ q, const u16* __restrict__ k,
            const u16* __restrict__ vt, const float* __restrict__ gm,
            const unsigned char* __restrict__ pad, u16* __restrict__ obf) {
  __shared__ float eS[16 * TLEN];
  __shared__ u16   aS[16 * TLEN];
  int qt = blockIdx.x, h = blockIdx.y, b = blockIdx.z;
  int bh = b * HEADS + h;
  const u16* Q  = q  + (size_t)bh * TLEN * HD;
  const u16* K  = k  + (size_t)bh * TLEN * HD;
  const u16* Vt = vt + (size_t)bh * HD * TLEN;   // [d][t]
  int l = threadIdx.x, lm = l & 15, kh = l >> 4;

  // phase 1: energy = (Q K^T) * gmask, masked  (Q frags hoisted)
  v16bf qf[3];
#pragma unroll
  for (int i = 0; i < 3; ++i) qf[i] = load_a_frag(Q, qt * 16, i * 32, HD);

  for (int nt = 0; nt < TLEN / 16; ++nt) {
    v8f acc = (v8f){0,0,0,0,0,0,0,0};
#pragma unroll
    for (int i = 0; i < 3; ++i) {
      v16bf bf = load_b_frag_t(K, nt * 16, i * 32, HD);   // B[d][n]=K[n,d]
      acc = wmma_bf16(qf[i], bf, acc);
    }
    int kg = nt * 16 + lm;
    bool padk = (kg < 2) || pad[b * NNODE + kg - 2];
#pragma unroll
    for (int r = 0; r < 8; ++r) {
      int qg = qt * 16 + r + 8 * kh;
      float g;
      if (qg < 2)      g = 1.0f;
      else if (kg < 2) g = 0.0f;
      else {
        int i = qg - 2, j = kg - 2;
        g = (i == j) ? 5.0f : gm[((size_t)b * NNODE + i) * NNODE + j];
      }
      bool padq = (qg < 2) || pad[b * NNODE + qg - 2];
      eS[(r + 8 * kh) * TLEN + kg] = (padq && padk) ? acc[r] * g : -1e20f;
    }
  }
  __syncthreads();

  // phase 2: softmax(energy / sqrt(EMB)); 2 lanes per row, 256 cols each
  const float invS = 1.0f / 27.712812921102035f;  // 1/sqrt(768)
  float mmax = -3.0e38f;
  for (int c = 0; c < 256; ++c) mmax = fmaxf(mmax, eS[lm * TLEN + kh * 256 + c]);
  mmax = fmaxf(mmax, __shfl_xor(mmax, 16));
  float ssum = 0.0f;
  for (int c = 0; c < 256; ++c) ssum += __expf((eS[lm * TLEN + kh * 256 + c] - mmax) * invS);
  ssum += __shfl_xor(ssum, 16);
  float rn = 1.0f / ssum;
  for (int c = 0; c < 256; ++c) {
    int idx = lm * TLEN + kh * 256 + c;
    aS[idx] = f2bf(__expf((eS[idx] - mmax) * invS) * rn);
  }
  __syncthreads();

  // phase 3: O = att @ V ; A from LDS, B from transposed V (contiguous in k)
  for (int dt = 0; dt < HD / 16; ++dt) {
    v8f acc = (v8f){0,0,0,0,0,0,0,0};
    for (int kt = 0; kt < TLEN; kt += 32) {
      v16bf af = load_a_frag(aS, 0, kt, TLEN);
      v16bf bf = load_b_frag_t(Vt, dt * 16, kt, TLEN);   // B[t][d] = Vt[d*TLEN+t]
      acc = wmma_bf16(af, bf, acc);
    }
    size_t row0 = (size_t)b * TLEN + qt * 16;
    store_acc_bf(obf, acc, row0, h * HD + dt * 16 + lm, EMB);
  }
}

// ---- generic WMMA GEMM with TDM double-buffered B staging ----------------
// block = 256 threads (8 waves), block tile 128x64.
// bT=1: B[k][n] = W[n*K+k];  bT=0: B[k][n] = W[k*N+n]
// outb!=null: out = bf16(relu(acc + bias));  else raw f32 to outf.

__global__ void __launch_bounds__(256)
gemm_kernel(const u16* __restrict__ A, const u16* __restrict__ Bm,
            int N, int K, int bT, const float* __restrict__ bias,
            float* __restrict__ outf, u16* __restrict__ outb) {
  __shared__ u16 bs[2][32 * 64];       // 4 KB per buffer
  int wave = threadIdx.x >> 5;
  int m0 = blockIdx.y * 128 + wave * 16;
  int n0 = blockIdx.x * 64;
  int nIter = K / 32;

  v8f acc[4];
#pragma unroll
  for (int j = 0; j < 4; ++j) acc[j] = (v8f){0,0,0,0,0,0,0,0};

  // TDM issue for K-slice `it` into buffer it&1 (wave 0 only)
  auto issue = [&](int it) {
    int kt = it * 32;
    if (bT) {   // 64 lines (n), 32 contiguous k each; pitch K
      tdm_load_2d(Bm + (size_t)n0 * K + kt, bs[it & 1],
                  (u32)K, (u32)N, 32u, 64u, (u64)K);
    } else {    // 32 lines (k), 64 contiguous n each; pitch N
      tdm_load_2d(Bm + (size_t)kt * N + n0, bs[it & 1],
                  (u32)N, (u32)K, 64u, 32u, (u64)N);
    }
  };

  if (wave == 0) issue(0);
  for (int it = 0; it < nIter; ++it) {
    if (wave == 0) {
      if (it + 1 < nIter) {
        issue(it + 1);
        __builtin_amdgcn_s_wait_tensorcnt(1);   // slice `it` landed in LDS
      } else {
        __builtin_amdgcn_s_wait_tensorcnt(0);
      }
    }
    __syncthreads();
    const u16* bp = bs[it & 1];
    v16bf af = load_a_frag(A, m0, it * 32, K);
#pragma unroll
    for (int j = 0; j < 4; ++j) {
      v16bf bf = bT ? load_b_frag_t(bp, 16 * j, 0, 32)
                    : load_b_frag_n(bp, 16 * j, 0, 64);
      acc[j] = wmma_bf16(af, bf, acc[j]);
    }
    __syncthreads();   // buffer it&1 free for reuse at iteration it+2
  }

  int lm = threadIdx.x & 15, kh = (threadIdx.x >> 4) & 1;
#pragma unroll
  for (int j = 0; j < 4; ++j) {
    int n = n0 + 16 * j + lm;
    if (outb) {
      float bs2 = bias ? bias[n] : 0.0f;
#pragma unroll
      for (int r = 0; r < 8; ++r) {
        float y = fmaxf(acc[j][r] + bs2, 0.0f);   // FFN1: bias + ReLU
        outb[(size_t)(m0 + r + 8 * kh) * N + n] = f2bf(y);
      }
    } else {
#pragma unroll
      for (int r = 0; r < 8; ++r)
        outf[(size_t)(m0 + r + 8 * kh) * N + n] = acc[j][r];
    }
  }
}

// ---- residual + bias + LayerNorm; one block per row ----------------------

__global__ void __launch_bounds__(256)
ln_kernel(const float* __restrict__ gin, const float* __restrict__ resid,
          const float* __restrict__ bias, const float* __restrict__ gamma,
          const float* __restrict__ beta, float* __restrict__ xout,
          u16* __restrict__ xbout) {
  __shared__ float red[256];
  size_t row = blockIdx.x;
  const float* gp = gin + row * EMB;
  const float* rp = resid + row * EMB;
  float vals[3]; float s = 0.0f;
#pragma unroll
  for (int i = 0; i < 3; ++i) {
    int c = threadIdx.x + 256 * i;
    float t = gp[c] + rp[c] + bias[c];
    vals[i] = t; s += t;
  }
  red[threadIdx.x] = s; __syncthreads();
  for (int st = 128; st > 0; st >>= 1) {
    if ((int)threadIdx.x < st) red[threadIdx.x] += red[threadIdx.x + st];
    __syncthreads();
  }
  float mean = red[0] * (1.0f / EMB); __syncthreads();
  float s2 = 0.0f;
#pragma unroll
  for (int i = 0; i < 3; ++i) { float d = vals[i] - mean; s2 += d * d; }
  red[threadIdx.x] = s2; __syncthreads();
  for (int st = 128; st > 0; st >>= 1) {
    if ((int)threadIdx.x < st) red[threadIdx.x] += red[threadIdx.x + st];
    __syncthreads();
  }
  float rstd = rsqrtf(red[0] * (1.0f / EMB) + 1e-5f);
#pragma unroll
  for (int i = 0; i < 3; ++i) {
    int c = threadIdx.x + 256 * i;
    float y = (vals[i] - mean) * rstd * gamma[c] + beta[c];
    xout[row * EMB + c] = y;
    xbout[row * EMB + c] = f2bf(y);
  }
}

// ---- final anchor projection ---------------------------------------------

__global__ void __launch_bounds__(256)
anchor_kernel(const float* __restrict__ xf, float* __restrict__ p) {
  __shared__ float anc[EMB];
  __shared__ float red[256];
  int b = blockIdx.x;
  const float* r0 = xf + (size_t)b * TLEN * EMB;
  const float* r1 = r0 + EMB;
  float s = 0.0f;
#pragma unroll
  for (int i = 0; i < 3; ++i) {
    int c = threadIdx.x + 256 * i;
    float a = r0[c] - r1[c];
    anc[c] = a; s += a * a;
  }
  red[threadIdx.x] = s; __syncthreads();
  for (int st = 128; st > 0; st >>= 1) {
    if ((int)threadIdx.x < st) red[threadIdx.x] += red[threadIdx.x + st];
    __syncthreads();
  }
  float inv = rsqrtf(red[0]); __syncthreads();
  for (int n = threadIdx.x; n < NNODE; n += 256) {
    const float* node = xf + ((size_t)b * TLEN + 2 + n) * EMB;
    float d = 0.0f;
    for (int c = 0; c < EMB; ++c) d += node[c] * anc[c];
    p[b * TLEN + n] = d * inv;
  }
}

__global__ void outA_kernel(const float* __restrict__ p, float* __restrict__ A) {
  size_t idx = (size_t)blockIdx.x * 256 + threadIdx.x;
  if (idx >= (size_t)BATCH * NNODE * NNODE) return;
  int j = (int)(idx % NNODE);
  size_t r = idx / NNODE;
  int i = (int)(r % NNODE);
  int b = (int)(r / NNODE);
  A[idx] = 2.0f * (p[b * TLEN + i] - p[b * TLEN + j]);
}

// ---- host orchestration --------------------------------------------------

extern "C" void kernel_launch(void* const* d_in, const int* in_sizes, int n_in,
                              void* d_out, int out_size, void* d_ws, size_t ws_size,
                              hipStream_t stream) {
  const float* embs = (const float*)d_in[0];
  const float* gm   = (const float*)d_in[1];
  const unsigned char* pad = (const unsigned char*)d_in[2];
  const float* Wq = (const float*)d_in[3];
  const float* Wk = (const float*)d_in[4];
  const float* Wv = (const float*)d_in[5];
  const float* Wo = (const float*)d_in[6];
  const float* bo = (const float*)d_in[7];
  const float* l1g = (const float*)d_in[8];
  const float* l1b = (const float*)d_in[9];
  const float* l2g = (const float*)d_in[10];
  const float* l2b = (const float*)d_in[11];
  const float* W1 = (const float*)d_in[12];
  const float* b1 = (const float*)d_in[13];
  const float* W2 = (const float*)d_in[14];
  const float* b2 = (const float*)d_in[15];
  const float* vs = (const float*)d_in[16];
  const float* vt = (const float*)d_in[17];
  float* Aout = (float*)d_out;

  char* base = (char*)d_ws;
  size_t off = 0;
  auto take = [&](size_t bytes) { char* p = base + off; off = (off + bytes + 255) & ~(size_t)255; return p; };

  const size_t actF = (size_t)MROWS * EMB * 4;
  const size_t actB = (size_t)MROWS * EMB * 2;
  float* xf  = (float*)take(actF);
  u16*   xb  = (u16*)take(actB);
  char*  qkvreg = take((size_t)MROWS * FFD * 2); // q,k,vT  OR  FFN hidden
  u16*   qb  = (u16*)qkvreg;
  u16*   kb  = (u16*)(qkvreg + actB);
  u16*   vtb = (u16*)(qkvreg + 2 * actB);
  u16*   fb  = (u16*)qkvreg;
  u16*   ob  = (u16*)take(actB);
  float* g32 = (float*)take(actF);
  float* x1f = (float*)take(actF);
  u16*   x1b = (u16*)take(actB);
  u16* wq = (u16*)take((size_t)NLAYER * HD * HD * 2);
  u16* wk = (u16*)take((size_t)NLAYER * HD * HD * 2);
  u16* wv = (u16*)take((size_t)NLAYER * HD * HD * 2);
  u16* wo = (u16*)take((size_t)NLAYER * EMB * EMB * 2);
  u16* w1 = (u16*)take((size_t)NLAYER * EMB * FFD * 2);
  u16* w2 = (u16*)take((size_t)NLAYER * FFD * EMB * 2);
  float* pbuf = (float*)take((size_t)BATCH * TLEN * 4);

  auto cvt = [&](const float* s, u16* d, int n) {
    cvt_bf16_kernel<<<(n + 255) / 256, 256, 0, stream>>>(s, d, n);
  };
  cvt(Wq, wq, NLAYER * HD * HD);
  cvt(Wk, wk, NLAYER * HD * HD);
  cvt(Wv, wv, NLAYER * HD * HD);
  cvt(Wo, wo, NLAYER * EMB * EMB);
  cvt(W1, w1, NLAYER * EMB * FFD);
  cvt(W2, w2, NLAYER * FFD * EMB);

  {
    size_t tot = (size_t)MROWS * EMB;
    build_x_kernel<<<(unsigned)((tot + 255) / 256), 256, 0, stream>>>(embs, vs, vt, xf, xb);
  }

  for (int L = 0; L < NLAYER; ++L) {
    qkv_kernel<<<dim3(TLEN / 16, BATCH * HEADS, 3), 32, 0, stream>>>(
        xb, wq + (size_t)L * HD * HD, wk + (size_t)L * HD * HD, wv + (size_t)L * HD * HD,
        qb, kb, vtb);
    attn_kernel<<<dim3(TLEN / 16, HEADS, BATCH), 32, 0, stream>>>(qb, kb, vtb, gm, pad, ob);
    gemm_kernel<<<dim3(EMB / 64, MROWS / 128), 256, 0, stream>>>(
        ob, wo + (size_t)L * EMB * EMB, EMB, EMB, 1, nullptr, g32, nullptr);
    ln_kernel<<<MROWS, 256, 0, stream>>>(g32, xf, bo + L * EMB, l1g + L * EMB, l1b + L * EMB, x1f, x1b);
    gemm_kernel<<<dim3(FFD / 64, MROWS / 128), 256, 0, stream>>>(
        x1b, w1 + (size_t)L * EMB * FFD, FFD, EMB, 0, b1 + L * FFD, nullptr, fb);
    gemm_kernel<<<dim3(EMB / 64, MROWS / 128), 256, 0, stream>>>(
        fb, w2 + (size_t)L * FFD * EMB, EMB, FFD, 0, nullptr, g32, nullptr);
    ln_kernel<<<MROWS, 256, 0, stream>>>(g32, x1f, b2 + L * EMB, l2g + L * EMB, l2b + L * EMB, xf, xb);
  }

  anchor_kernel<<<BATCH, 256, 0, stream>>>(xf, pbuf);
  {
    size_t tot = (size_t)BATCH * NNODE * NNODE;
    outA_kernel<<<(unsigned)((tot + 255) / 256), 256, 0, stream>>>(pbuf, Aout);
  }
}
